// H2GCN_24481313587843
// MI455X (gfx1250) — matrix operand order repaired and checked by reading, not compiled, not run on hardware
//
#include <hip/hip_runtime.h>
#include <hip/hip_bf16.h>

typedef __attribute__((ext_vector_type(16))) __bf16 v16bf;
typedef __attribute__((ext_vector_type(8)))  float  v8f;

#define IN_C  128
#define HID   128
#define OUT_C 40
#define LDWC  48   // padded leading dim for Wc bf16 copy
#define KCAT  384

// ---------------------------------------------------------------------------
// Small elementwise / setup kernels
// ---------------------------------------------------------------------------
__global__ void h2gcn_zero_f32(float* __restrict__ p, long long n) {
    long long i = (long long)blockIdx.x * blockDim.x + threadIdx.x;
    if (i < n) p[i] = 0.0f;
}

// agg[i] = bias[i % 128]  (pre-load bias so scatter-add lands on top of it)
__global__ void h2gcn_fill_bias(float* __restrict__ agg, const float* __restrict__ b,
                                long long n) {
    long long i = (long long)blockIdx.x * blockDim.x + threadIdx.x;
    if (i < n) agg[i] = b[i & (HID - 1)];
}

__global__ void h2gcn_cvt_bf16(const float* __restrict__ src, __bf16* __restrict__ dst,
                               long long n) {
    long long i = (long long)blockIdx.x * blockDim.x + threadIdx.x;
    if (i < n) dst[i] = (__bf16)src[i];
}

// Wc [384 x 40] f32  ->  bf16 padded [384 x 48] (zeros in pad)
__global__ void h2gcn_cvt_wc(const float* __restrict__ wc, __bf16* __restrict__ dst) {
    int i = blockIdx.x * blockDim.x + threadIdx.x;
    if (i >= KCAT * LDWC) return;
    int k = i / LDWC, n = i % LDWC;
    dst[i] = (n < OUT_C) ? (__bf16)wc[k * OUT_C + n] : (__bf16)0.0f;
}

// bc [40] -> padded f32 [48] (zeros in pad) so the final-GEMM bias load is unguarded
__global__ void h2gcn_pad_bc(const float* __restrict__ bc, float* __restrict__ dst) {
    int i = blockIdx.x * blockDim.x + threadIdx.x;
    if (i < LDWC) dst[i] = (i < OUT_C) ? bc[i] : 0.0f;
}

// deg[col[e]] += 1
__global__ void h2gcn_deg(const long long* __restrict__ col, float* __restrict__ deg,
                          long long E) {
    long long e = (long long)blockIdx.x * blockDim.x + threadIdx.x;
    if (e < E) atomicAdd(&deg[(int)col[e]], 1.0f);
}

__global__ void h2gcn_dinv(const float* __restrict__ deg, float* __restrict__ dinv, int n) {
    int i = blockIdx.x * blockDim.x + threadIdx.x;
    if (i >= n) return;
    float d = deg[i];
    dinv[i] = (d > 0.0f) ? rsqrtf(fmaxf(d, 1.0f)) : 0.0f;
}

// agg[col[e]][:] += dinv[row]*dinv[col] * src[row[e]][:]
// one wave per edge; lane q handles channels [4q, 4q+4)
__global__ void h2gcn_scatter(const long long* __restrict__ row,
                              const long long* __restrict__ col,
                              const float* __restrict__ dinv,
                              const float* __restrict__ src,
                              float* __restrict__ agg, long long E) {
    long long gid = (long long)blockIdx.x * blockDim.x + threadIdx.x;
    long long e = gid >> 5;
    if (e >= E) return;
    int q = (int)(gid & 31);
    int r = (int)row[e];
    int c = (int)col[e];
    float norm = dinv[r] * dinv[c];
    const float4 v = *(const float4*)(src + (size_t)r * HID + q * 4);
    float* dst = agg + (size_t)c * HID + q * 4;
    atomicAdd(dst + 0, norm * v.x);
    atomicAdd(dst + 1, norm * v.y);
    atomicAdd(dst + 2, norm * v.z);
    atomicAdd(dst + 3, norm * v.w);
}

// ---------------------------------------------------------------------------
// WMMA fragment helpers (ISA 7.12.2 layouts, wave32)
// A fragment, 16x32 bf16, from row-major f32 A (lda = cols), rows rowBase..+15.
// Row index clamped to M-1: keeps EXEC all-1s (WMMA requirement) and keeps
// the ragged last 32-row strip in-bounds when reading the exact-sized input x.
// ---------------------------------------------------------------------------
template <bool RELUA>
__device__ __forceinline__ v16bf h2gcn_load_a(const float* __restrict__ A, int rowBase,
                                              int lda, int kb, int lane, int M) {
    int half = lane >> 4;          // 0: K in {kb..+7, kb+16..+23}; 1: +8
    int m    = lane & 15;          // row within tile
    int r    = min(rowBase + m, M - 1);
    const float* p = A + (size_t)r * lda + kb + 8 * half;
    v16bf a;
#pragma unroll
    for (int i = 0; i < 8; ++i) {
        float x0 = p[i];
        float x1 = p[16 + i];
        if (RELUA) { x0 = fmaxf(x0, 0.0f); x1 = fmaxf(x1, 0.0f); }
        a[i]     = (__bf16)x0;     // K = kb + 8*half + i
        a[8 + i] = (__bf16)x1;     // K = kb + 16 + 8*half + i
    }
    return a;
}

// ---------------------------------------------------------------------------
// GEMM:  C[Mpad x 128] = maybe_relu( maybe_relu_in(A[M x 128]) @ Wbf[128x128] + bias )
// One wave per 32-row strip (two 16-row A fragments share every B fragment).
// C is a ws buffer padded to a 32-row multiple -> stores need NO guards.
// M is assumed to be a multiple of 16 (N = 50000 = 3125*16).
// ---------------------------------------------------------------------------
template <bool RELUA, bool BIAS, bool RELUOUT>
__global__ void h2gcn_gemm128(const float* __restrict__ A, const __bf16* __restrict__ Wbf,
                              const float* __restrict__ bias, float* __restrict__ C,
                              int M) {
    int wave = threadIdx.x >> 5;
    int lane = threadIdx.x & 31;
    int rowBase = (blockIdx.x * (blockDim.x >> 5) + wave) * 32;
    if (rowBase >= M) return;
    int m = lane & 15, half = lane >> 4;

    v8f acc0[8] = {};
    v8f acc1[8] = {};
    for (int kb = 0; kb < IN_C; kb += 32) {
        v16bf a0 = h2gcn_load_a<RELUA>(A, rowBase,      IN_C, kb, lane, M);
        v16bf a1 = h2gcn_load_a<RELUA>(A, rowBase + 16, IN_C, kb, lane, M);
        const __bf16* brow = Wbf + (size_t)(kb + 8 * half) * HID + m;
#pragma unroll
        for (int ct = 0; ct < 8; ++ct) {
            v16bf b;
#pragma unroll
            for (int i = 0; i < 8; ++i) {
                b[i]     = brow[(size_t)i * HID + ct * 16];
                b[8 + i] = brow[(size_t)(16 + i) * HID + ct * 16];
            }
            acc0[ct] = __builtin_amdgcn_wmma_f32_16x16x32_bf16(
                false, a0, false, b, (short)0, acc0[ct], false, false);
            acc1[ct] = __builtin_amdgcn_wmma_f32_16x16x32_bf16(
                false, a1, false, b, (short)0, acc1[ct], false, false);
        }
    }
#pragma unroll
    for (int s = 0; s < 2; ++s) {
        v8f* acc = s ? acc1 : acc0;
        float* crow = C + (size_t)(rowBase + s * 16 + 8 * half) * HID + m;
#pragma unroll
        for (int ct = 0; ct < 8; ++ct) {
            float bn = BIAS ? bias[ct * 16 + m] : 0.0f;
#pragma unroll
            for (int r = 0; r < 8; ++r) {
                float v = acc[ct][r] + bn;
                if (RELUOUT) v = fmaxf(v, 0.0f);
                crow[(size_t)r * HID + ct * 16] = v;
            }
        }
    }
}

// ---------------------------------------------------------------------------
// Final GEMM: out[M x 40] = relu-on-load(concat(hego, agg1, agg2)) @ Wc + bcp
// Three sequential 128-K phases with compile-time A pointers (no branchy
// per-iteration pointer select). Wc bf16 padded [384 x 48]; bcp padded [48].
// out is exact-sized: second strip of the last wave is skipped (wave-uniform,
// M % 16 == 0), lane-level predicate only on n < 40.
// ---------------------------------------------------------------------------
__global__ void h2gcn_gemm_final(const float* __restrict__ hego,
                                 const float* __restrict__ agg1,
                                 const float* __restrict__ agg2,
                                 const __bf16* __restrict__ Wcbf,
                                 const float* __restrict__ bcp,
                                 float* __restrict__ out, int M) {
    int wave = threadIdx.x >> 5;
    int lane = threadIdx.x & 31;
    int rowBase = (blockIdx.x * (blockDim.x >> 5) + wave) * 32;
    if (rowBase >= M) return;
    int m = lane & 15, half = lane >> 4;

    v8f acc0[3] = {};
    v8f acc1[3] = {};

#pragma unroll
    for (int ph = 0; ph < 3; ++ph) {
        const float* A = (ph == 0) ? hego : (ph == 1) ? agg1 : agg2;
        for (int kb = 0; kb < HID; kb += 32) {
            // relu on load: idempotent on hego, realizes h1/h2 from agg1/agg2
            v16bf a0 = h2gcn_load_a<true>(A, rowBase,      HID, kb, lane, M);
            v16bf a1 = h2gcn_load_a<true>(A, rowBase + 16, HID, kb, lane, M);
            int k = ph * HID + kb + 8 * half;
            const __bf16* brow = Wcbf + (size_t)k * LDWC + m;
#pragma unroll
            for (int ct = 0; ct < 3; ++ct) {
                v16bf b;
#pragma unroll
                for (int i = 0; i < 8; ++i) {
                    b[i]     = brow[(size_t)i * LDWC + ct * 16];
                    b[8 + i] = brow[(size_t)(16 + i) * LDWC + ct * 16];
                }
                acc0[ct] = __builtin_amdgcn_wmma_f32_16x16x32_bf16(
                    false, a0, false, b, (short)0, acc0[ct], false, false);
                acc1[ct] = __builtin_amdgcn_wmma_f32_16x16x32_bf16(
                    false, a1, false, b, (short)0, acc1[ct], false, false);
            }
        }
    }
#pragma unroll
    for (int s = 0; s < 2; ++s) {
        int sBase = rowBase + s * 16;
        if (sBase >= M) break;                         // wave-uniform (M % 16 == 0)
        v8f* acc = s ? acc1 : acc0;
        float* orow = out + (size_t)(sBase + 8 * half) * OUT_C;
#pragma unroll
        for (int ct = 0; ct < 3; ++ct) {
            int n = ct * 16 + m;
            float bn = bcp[n];                         // padded, unguarded load
            bool ok = (n < OUT_C);
#pragma unroll
            for (int r = 0; r < 8; ++r) {
                if (ok) orow[(size_t)r * OUT_C + n] = acc[ct][r] + bn;
            }
        }
    }
}

// ---------------------------------------------------------------------------
// Host launcher
// ---------------------------------------------------------------------------
extern "C" void kernel_launch(void* const* d_in, const int* in_sizes, int n_in,
                              void* d_out, int out_size, void* d_ws, size_t ws_size,
                              hipStream_t stream) {
    const float*      x     = (const float*)d_in[0];
    const long long*  ei    = (const long long*)d_in[1];
    const float*      W_ego = (const float*)d_in[2];
    const float*      b_ego = (const float*)d_in[3];
    const float*      W1    = (const float*)d_in[4];
    const float*      b1    = (const float*)d_in[5];
    const float*      W2    = (const float*)d_in[6];
    const float*      b2    = (const float*)d_in[7];
    const float*      Wc    = (const float*)d_in[8];
    const float*      bc    = (const float*)d_in[9];
    float* out = (float*)d_out;

    const int       N    = in_sizes[0] / IN_C;         // 50000 (multiple of 16)
    const int       Npad = (N + 31) & ~31;             // padded rows for ws buffers
    const long long E    = (long long)in_sizes[1] / 2; // 600000
    const long long* row = ei;
    const long long* col = ei + E;

    // workspace carve-out (256B aligned)
    char* ws = (char*)d_ws;
    size_t off = 0;
    auto carve = [&](size_t bytes) -> char* {
        char* p = ws + off;
        off = (off + bytes + 255) & ~(size_t)255;
        return p;
    };
    float*  deg   = (float*)carve((size_t)N * sizeof(float));
    float*  dinv  = (float*)carve((size_t)N * sizeof(float));
    float*  hego  = (float*)carve((size_t)Npad * HID * sizeof(float));
    float*  tmp   = (float*)carve((size_t)Npad * HID * sizeof(float)); // xW1, then h1W2
    float*  agg1  = (float*)carve((size_t)Npad * HID * sizeof(float));
    float*  agg2  = (float*)carve((size_t)Npad * HID * sizeof(float));
    __bf16* wbe   = (__bf16*)carve((size_t)IN_C * HID * sizeof(__bf16));
    __bf16* wb1   = (__bf16*)carve((size_t)IN_C * HID * sizeof(__bf16));
    __bf16* wb2   = (__bf16*)carve((size_t)HID * HID * sizeof(__bf16));
    __bf16* wcb   = (__bf16*)carve((size_t)KCAT * LDWC * sizeof(__bf16));
    float*  bcp   = (float*)carve((size_t)LDWC * sizeof(float));
    (void)ws_size; (void)n_in; (void)out_size;

    const int T = 256;
    const long long NH = (long long)N * HID;

    // init (every call: deterministic, graph-capture safe)
    h2gcn_zero_f32 <<<(N + T - 1) / T, T, 0, stream>>>(deg, N);
    h2gcn_fill_bias<<<(unsigned)((NH + T - 1) / T), T, 0, stream>>>(agg1, b1, NH);
    h2gcn_fill_bias<<<(unsigned)((NH + T - 1) / T), T, 0, stream>>>(agg2, b2, NH);
    h2gcn_cvt_bf16 <<<(IN_C * HID + T - 1) / T, T, 0, stream>>>(W_ego, wbe, IN_C * HID);
    h2gcn_cvt_bf16 <<<(IN_C * HID + T - 1) / T, T, 0, stream>>>(W1,    wb1, IN_C * HID);
    h2gcn_cvt_bf16 <<<(HID  * HID + T - 1) / T, T, 0, stream>>>(W2,    wb2, HID * HID);
    h2gcn_cvt_wc   <<<(KCAT * LDWC + T - 1) / T, T, 0, stream>>>(Wc, wcb);
    h2gcn_pad_bc   <<<1, 64, 0, stream>>>(bc, bcp);

    // degree + symmetric norm
    h2gcn_deg <<<(unsigned)((E + T - 1) / T), T, 0, stream>>>(col, deg, E);
    h2gcn_dinv<<<(N + T - 1) / T, T, 0, stream>>>(deg, dinv, N);

    // GEMMs: 8 waves/block, each wave -> 32-row strip
    const int strips  = Npad / 32;
    const int gBlocks = (strips + 7) / 8;
    h2gcn_gemm128<false, true,  true ><<<gBlocks, 256, 0, stream>>>(x, wbe, b_ego, hego, N);
    h2gcn_gemm128<false, false, false><<<gBlocks, 256, 0, stream>>>(x, wb1, nullptr, tmp, N);

    // conv1 aggregation: agg1 (pre-loaded with b1) += norm * tmp[row]
    const long long sThreads = E * 32;
    h2gcn_scatter<<<(unsigned)((sThreads + T - 1) / T), T, 0, stream>>>(row, col, dinv,
                                                                       tmp, agg1, E);

    // h1@W2 with relu-on-load of agg1
    h2gcn_gemm128<true, false, false><<<gBlocks, 256, 0, stream>>>(agg1, wb2, nullptr, tmp, N);

    // conv2 aggregation
    h2gcn_scatter<<<(unsigned)((sThreads + T - 1) / T), T, 0, stream>>>(row, col, dinv,
                                                                       tmp, agg2, E);

    // classifier: concat(hego, relu(agg1), relu(agg2)) @ Wc + bcp
    h2gcn_gemm_final<<<gBlocks, 256, 0, stream>>>(hego, agg1, agg2, wcb, bcp, out, N);
}
